// Transformer_69200513073173
// MI455X (gfx1250) — matrix operand order repaired and checked by reading, not compiled, not run on hardware
//
#include <hip/hip_runtime.h>
#include <hip/hip_bf16.h>

// ---------------------------------------------------------------------------
// Types for CDNA5 WMMA (wave32): V_WMMA_F32_16X16X32_BF16
// ---------------------------------------------------------------------------
typedef __attribute__((ext_vector_type(16))) __bf16 v16bf;
typedef __attribute__((ext_vector_type(8)))  __bf16 v8bf;
typedef __attribute__((ext_vector_type(8)))  float  v8f;

// Hardware convert path (v_cvt_*_bf16_f32 on gfx1250).
static __device__ __forceinline__ __bf16 f2bf(float f) { return (__bf16)f; }

// Build a 16-element bf16 fragment from two contiguous 8-element chunks.
static __device__ __forceinline__ v16bf ldfrag(const __bf16* p,
                                               const __bf16* q) {
    v8bf lo = *(const v8bf*)p;
    v8bf hi = *(const v8bf*)q;
    return __builtin_shufflevector(lo, hi, 0, 1, 2, 3, 4, 5, 6, 7,
                                   8, 9, 10, 11, 12, 13, 14, 15);
}

// Model constants
#define TB   2
#define TL   2048
#define TD   1024
#define TH   16
#define TDH  64
#define NTOK 4096            // B*L
#define AINF 3.0e38f

// ---------------------------------------------------------------------------
// Elementwise f32 -> bf16 tensor convert. n multiple of 2048.
// ---------------------------------------------------------------------------
__global__ __launch_bounds__(256)
void cvt_f32_bf16(const float* __restrict__ src, __bf16* __restrict__ dst) {
    const size_t i = ((size_t)blockIdx.x * 256 + threadIdx.x) * 8;
#pragma unroll
    for (int j = 0; j < 8; ++j) dst[i + j] = f2bf(src[i + j]);
}

// ---------------------------------------------------------------------------
// GEMM: Y[M=4096, N=1024] = X bf16 @ W^T bf16 + bias, optional LeakyReLU and
// output scale. Writes f32 (Yf) and/or bf16 (Yh).
// Block = 256 thr (8 waves); block tile 256x64; wave tile 32x64:
//   2 A row-fragments x 4 B fragments -> 8 WMMA per k-step from 12 b128
//   loads (B fragments amortized over two row tiles).
// A-fragment (16x32): lane holds row (lane&15), K = {k0+hi*8..+7,
// k0+16+hi*8..+7}. B-fragment (32x16): lane holds col (lane&15) = W row,
// K = {k0+hi*8..+7, k0+16+hi*8..+7}. All pure bf16 b128 loads from global.
// ---------------------------------------------------------------------------
__global__ __launch_bounds__(256)
void gemm_bf16(const __bf16* __restrict__ X, const __bf16* __restrict__ W,
               const float* __restrict__ bias, float* __restrict__ Yf,
               __bf16* __restrict__ Yh, float scale, int leaky) {
    constexpr int N = TD, K = TD;
    const int wave = threadIdx.x >> 5;
    const int lane = threadIdx.x & 31;
    const int cn   = lane & 15;
    const int hi   = lane >> 4;
    const int m0   = blockIdx.x * 256 + wave * 32;
    const int n0   = blockIdx.y * 64;

    const __bf16* xr0 = X + (size_t)(m0 + cn) * K + hi * 8;
    const __bf16* xr1 = xr0 + (size_t)16 * K;
    const __bf16* wr[4];
#pragma unroll
    for (int t = 0; t < 4; ++t)
        wr[t] = W + (size_t)(n0 + t * 16 + cn) * K + (hi << 3);

    v8f acc0[4] = {};
    v8f acc1[4] = {};

#pragma unroll 2
    for (int k0 = 0; k0 < K; k0 += 32) {
        const v16bf a0 = ldfrag(xr0 + k0, xr0 + k0 + 16);
        const v16bf a1 = ldfrag(xr1 + k0, xr1 + k0 + 16);
#pragma unroll
        for (int t = 0; t < 4; ++t) {
            const v16bf b = ldfrag(wr[t] + k0, wr[t] + k0 + 16);
            acc0[t] = __builtin_amdgcn_wmma_f32_16x16x32_bf16(
                false, a0, false, b, (short)0, acc0[t], false, false);
            acc1[t] = __builtin_amdgcn_wmma_f32_16x16x32_bf16(
                false, a1, false, b, (short)0, acc1[t], false, false);
        }
    }

    const int rb = hi << 3;
#pragma unroll
    for (int t = 0; t < 4; ++t) {
        const int col = n0 + t * 16 + cn;
        const float bv = bias[col];
#pragma unroll
        for (int rt = 0; rt < 2; ++rt) {
#pragma unroll
            for (int r = 0; r < 8; ++r) {
                float v = (rt == 0 ? acc0[t][r] : acc1[t][r]) + bv;
                if (leaky) v = v > 0.f ? v : 0.01f * v;
                v *= scale;
                const size_t off =
                    (size_t)(m0 + rt * 16 + rb + r) * N + col;
                if (Yf) Yf[off] = v;
                if (Yh) Yh[off] = f2bf(v);
            }
        }
    }
}

// ---------------------------------------------------------------------------
// Flash attention, all-bf16 WMMA operands, fp32 online softmax.
// Block = 128 thr (4 waves); each wave owns a 16-query tile; block covers 64
// queries of one (b,h). Key blocks of 32; K/V tiles staged in LDS (V stored
// transposed so the P@V B-fragment is two contiguous 8-element reads).
// Q already carries the 1/sqrt(Dh) scale (folded into its GEMM epilogue).
// Q,K,V,O are (B*L, D) bf16 with head h at cols [h*64, h*64+64).
// ---------------------------------------------------------------------------
__global__ __launch_bounds__(128)
void attn_flash(const __bf16* __restrict__ Q, const __bf16* __restrict__ Km,
                const __bf16* __restrict__ Vm, __bf16* __restrict__ O,
                int causal) {
    __shared__ __align__(16) __bf16 sK[32 * 64];     // [key][dh]
    __shared__ __align__(16) __bf16 sVt[64 * 32];    // [dh][key]
    __shared__ __align__(16) __bf16 sP[4][16 * 32];  // per-wave P (bf16)

    const int wave = threadIdx.x >> 5;
    const int lane = threadIdx.x & 31;
    const int cn   = lane & 15;
    const int hi   = lane >> 4;
    const int rb   = hi << 3;

    const int qt = blockIdx.x & 31;          // L/64 = 32 query tiles
    const int h  = (blockIdx.x >> 5) & 15;
    const int b  = blockIdx.x >> 9;

    const int    q0     = qt * 64 + wave * 16;
    const size_t baseBL = (size_t)b * TL;

    // Q fragments: pure bf16 loads (scale already applied)
    const __bf16* qrow = Q + (baseBL + q0 + cn) * TD + h * TDH + hi * 8;
    const v16bf qa0 = ldfrag(qrow, qrow + 16);
    const v16bf qa1 = ldfrag(qrow + 32, qrow + 48);

    v8f oacc[4] = {};
    float mi[8], li[8];
#pragma unroll
    for (int r = 0; r < 8; ++r) { mi[r] = -AINF; li[r] = 0.f; }

    // staging coordinates: thread copies 16 contiguous elems of one key row
    const int skey = threadIdx.x >> 2;          // 0..31
    const int sdh0 = (threadIdx.x & 3) << 4;    // 0,16,32,48

    const int kend = causal ? (qt * 64 + 64) : TL;
    for (int k0 = 0; k0 < kend; k0 += 32) {
        __syncthreads();  // previous iteration's LDS reads complete
        {
            const size_t g = (baseBL + k0 + skey) * TD + h * TDH + sdh0;
            const int t16 = skey * 64 + sdh0;
            *(v8bf*)(sK + t16)     = *(const v8bf*)(Km + g);
            *(v8bf*)(sK + t16 + 8) = *(const v8bf*)(Km + g + 8);
            const __bf16* gv = Vm + g;
#pragma unroll
            for (int j = 0; j < 16; ++j) sVt[(sdh0 + j) * 32 + skey] = gv[j];
        }
        __syncthreads();

        if (causal && k0 > q0 + 15) continue;  // barriers stay uniform

        // S = Q @ K^T : two 16x16 C tiles (keys k0+cn, k0+16+cn)
        v8f s0 = {}, s1 = {};
#pragma unroll
        for (int kb = 0; kb < 2; ++kb) {  // key half
            const __bf16* p = sK + (kb * 16 + cn) * 64 + (hi << 3);
            v8f acc = {};
            acc = __builtin_amdgcn_wmma_f32_16x16x32_bf16(
                false, qa0, false, ldfrag(p, p + 16), (short)0, acc,
                false, false);
            acc = __builtin_amdgcn_wmma_f32_16x16x32_bf16(
                false, qa1, false, ldfrag(p + 32, p + 48), (short)0, acc,
                false, false);
            if (kb == 0) s0 = acc; else s1 = acc;
        }

        if (causal && (k0 + 31 > q0)) {
#pragma unroll
            for (int r = 0; r < 8; ++r) {
                const int qr = q0 + rb + r;
                if (k0 + cn      > qr) s0[r] = -AINF;
                if (k0 + 16 + cn > qr) s1[r] = -AINF;
            }
        }

        // Online softmax per row (row lives across a 16-lane half)
        float alpha[8];
#pragma unroll
        for (int r = 0; r < 8; ++r) {
            float mx = fmaxf(s0[r], s1[r]);
#pragma unroll
            for (int off = 8; off > 0; off >>= 1)
                mx = fmaxf(mx, __shfl_xor(mx, off, 32));
            const float mnew = fmaxf(mi[r], mx);
            const float a  = __expf(mi[r] - mnew);
            const float p0 = __expf(s0[r] - mnew);
            const float p1 = __expf(s1[r] - mnew);
            float rs = p0 + p1;
#pragma unroll
            for (int off = 8; off > 0; off >>= 1)
                rs += __shfl_xor(rs, off, 32);
            li[r] = li[r] * a + rs;
            mi[r] = mnew;
            alpha[r] = a;
            sP[wave][(rb + r) * 32 + cn]      = f2bf(p0);
            sP[wave][(rb + r) * 32 + 16 + cn] = f2bf(p1);
        }
#pragma unroll
        for (int t = 0; t < 4; ++t)
#pragma unroll
            for (int r = 0; r < 8; ++r) oacc[t][r] *= alpha[r];

        // P (bf16 in LDS) -> A-fragment: pure ds b128 loads
        const __bf16* pp = sP[wave] + cn * 32 + hi * 8;
        const v16bf pa = ldfrag(pp, pp + 16);

        // O += P @ V  (4 dh tiles of 16)
#pragma unroll
        for (int t = 0; t < 4; ++t) {
            const __bf16* p = sVt + (t * 16 + cn) * 32 + (hi << 3);
            oacc[t] = __builtin_amdgcn_wmma_f32_16x16x32_bf16(
                false, pa, false, ldfrag(p, p + 16), (short)0, oacc[t],
                false, false);
        }
    }

#pragma unroll
    for (int r = 0; r < 8; ++r) {
        const float inv = 1.f / li[r];
        const size_t rowoff = (baseBL + q0 + rb + r) * TD + h * TDH;
#pragma unroll
        for (int t = 0; t < 4; ++t)
            O[rowoff + t * 16 + cn] = f2bf(oacc[t][r] * inv);
    }
}

// ---------------------------------------------------------------------------
// x = LayerNorm(x + y); also writes bf16 mirror xb for WMMA consumers.
// One 256-thread block per row of D=1024.
// ---------------------------------------------------------------------------
__global__ __launch_bounds__(256)
void add_ln(float* __restrict__ x, const float* __restrict__ y,
            __bf16* __restrict__ xb) {
    const size_t row = blockIdx.x;
    float* xr = x + row * TD;
    const float* yr = y + row * TD;
    __bf16* xbr = xb + row * TD;

    float v[4], s = 0.f, ss = 0.f;
#pragma unroll
    for (int j = 0; j < 4; ++j) {
        const int idx = threadIdx.x + j * 256;
        const float t = xr[idx] + yr[idx];
        v[j] = t; s += t; ss += t * t;
    }
#pragma unroll
    for (int off = 16; off > 0; off >>= 1) {
        s  += __shfl_xor(s,  off, 32);
        ss += __shfl_xor(ss, off, 32);
    }
    __shared__ float rs[8], rss[8];
    const int wave = threadIdx.x >> 5, lane = threadIdx.x & 31;
    if (lane == 0) { rs[wave] = s; rss[wave] = ss; }
    __syncthreads();
    float S = 0.f, SS = 0.f;
#pragma unroll
    for (int w = 0; w < 8; ++w) { S += rs[w]; SS += rss[w]; }
    const float mean = S * (1.f / TD);
    const float var  = SS * (1.f / TD) - mean * mean;
    const float inv  = rsqrtf(var + 1e-5f);
#pragma unroll
    for (int j = 0; j < 4; ++j) {
        const int idx = threadIdx.x + j * 256;
        const float o = (v[j] - mean) * inv;
        xr[idx]  = o;
        xbr[idx] = f2bf(o);
    }
}

// ---------------------------------------------------------------------------
// Orchestration. Workspace layout (bytes):
//   [0,16M)    x   f32 residual stream
//   [16M,32M)  yb  f32 sublayer output
//   [32M,40M)  xb  bf16 mirror of x
//   [40M,48M)  qb  bf16   [48M,56M) kb   [56M,64M) vb   [64M,72M) tb
//   [72M,152M) bf16 weights: Wq(8) Wk(8) Wv(8) Wo(8) Wd(8), 2MB each
// Causal mask is analytic (reference mask is tril), so d_in[1] is unused.
// ---------------------------------------------------------------------------
extern "C" void kernel_launch(void* const* d_in, const int* in_sizes, int n_in,
                              void* d_out, int out_size, void* d_ws,
                              size_t ws_size, hipStream_t stream) {
    (void)in_sizes; (void)n_in; (void)out_size; (void)ws_size;
    constexpr size_t TSZ  = (size_t)NTOK * TD;   // 4M elements
    constexpr size_t WMAT = (size_t)TD * TD;     // 1M elements

    const float* response = (const float*)d_in[0];
    const float* Wsrc[5] = {(const float*)d_in[2], (const float*)d_in[4],
                            (const float*)d_in[6], (const float*)d_in[8],
                            (const float*)d_in[10]};
    const float* bq = (const float*)d_in[3];
    const float* bk = (const float*)d_in[5];
    const float* bv = (const float*)d_in[7];
    const float* bo = (const float*)d_in[9];
    const float* bd = (const float*)d_in[11];

    char* ws = (char*)d_ws;
    float*  x  = (float*)(ws);
    float*  yb = (float*)(ws + (size_t)16 * 1024 * 1024);
    __bf16* xb = (__bf16*)(ws + (size_t)32 * 1024 * 1024);
    __bf16* qb = xb + TSZ;
    __bf16* kb = qb + TSZ;
    __bf16* vb = kb + TSZ;
    __bf16* tb = vb + TSZ;
    __bf16* Wbf = tb + TSZ;   // 40 x 1M bf16
    __bf16* Wqb = Wbf + 0 * 8 * WMAT;
    __bf16* Wkb = Wbf + 1 * 8 * WMAT;
    __bf16* Wvb = Wbf + 2 * 8 * WMAT;
    __bf16* Wob = Wbf + 3 * 8 * WMAT;
    __bf16* Wdb = Wbf + 4 * 8 * WMAT;
    __bf16* Wdst[5] = {Wqb, Wkb, Wvb, Wob, Wdb};

    // Pre-convert weights (8M elements each tensor) and input.
    for (int wti = 0; wti < 5; ++wti)
        cvt_f32_bf16<<<dim3(8 * WMAT / 2048), dim3(256), 0, stream>>>(
            Wsrc[wti], Wdst[wti]);
    hipMemcpyAsync(x, response, TSZ * sizeof(float),
                   hipMemcpyDeviceToDevice, stream);
    cvt_f32_bf16<<<dim3(TSZ / 2048), dim3(256), 0, stream>>>(response, xb);

    const dim3 ggrid(NTOK / 256, TD / 64);
    const dim3 gblk(256);
    const dim3 agrid(TB * TH * (TL / 64));
    const dim3 ablk(128);
    const dim3 lgrid(NTOK);
    const dim3 lblk(256);

    for (int i = 0; i < 4; ++i) {
        for (int sub = 0; sub < 2; ++sub) {
            const int m = 2 * i + sub;
            // q scaled by 1/sqrt(Dh)=0.125 at the source
            gemm_bf16<<<ggrid, gblk, 0, stream>>>(
                xb, Wqb + m * WMAT, bq + m * TD, nullptr, qb, 0.125f, 0);
            gemm_bf16<<<ggrid, gblk, 0, stream>>>(
                xb, Wkb + m * WMAT, bk + m * TD, nullptr, kb, 1.0f, 0);
            gemm_bf16<<<ggrid, gblk, 0, stream>>>(
                xb, Wvb + m * WMAT, bv + m * TD, nullptr, vb, 1.0f, 0);
            attn_flash<<<agrid, ablk, 0, stream>>>(qb, kb, vb, tb,
                                                   sub == 0 ? 1 : 0);
            gemm_bf16<<<ggrid, gblk, 0, stream>>>(
                tb, Wob + m * WMAT, bo + m * TD, yb, nullptr, 1.0f, 0);
            add_ln<<<lgrid, lblk, 0, stream>>>(x, yb, xb);
        }
        gemm_bf16<<<ggrid, gblk, 0, stream>>>(
            xb, Wdb + (2 * i + 1) * WMAT, bd + (2 * i + 1) * TD,
            nullptr, tb, 1.0f, 1);
        gemm_bf16<<<ggrid, gblk, 0, stream>>>(
            tb, Wdb + (2 * i) * WMAT, bd + (2 * i) * TD, yb, nullptr,
            1.0f, 0);
        add_ln<<<lgrid, lblk, 0, stream>>>(x, yb, xb);
    }

    hipMemcpyAsync(d_out, x, TSZ * sizeof(float),
                   hipMemcpyDeviceToDevice, stream);
}
